// SelfAttention_v1_13460427505739
// MI455X (gfx1250) — matrix operand order repaired and checked by reading, not compiled, not run on hardware
//
#include <hip/hip_runtime.h>
#include <hip/hip_bf16.h>

#define SEQ 4096
#define DIM 1024

typedef __attribute__((ext_vector_type(16))) __bf16       v16bf;
typedef __attribute__((ext_vector_type(8)))  float        v8f;
typedef __attribute__((ext_vector_type(4)))  unsigned int v4u;
typedef __attribute__((ext_vector_type(8)))  unsigned int v8u;
typedef __attribute__((ext_vector_type(4)))  int          v4i;

static __device__ inline v8f vzero8() {
    v8f z; for (int e = 0; e < 8; ++e) z[e] = 0.f; return z;
}

// 16 bf16 from one contiguous 32-byte chunk (B fragments: lane = K row, 16 N elems)
static __device__ inline v16bf load_frag1(const __bf16* p) {
    v8u c = *(const v8u*)p;
    return __builtin_bit_cast(v16bf, c);
}

// 16 bf16 from two 16-byte chunks (A fragments: K groups {0-7,16-23} or {8-15,24-31})
static __device__ inline v16bf load_frag2(const __bf16* p0, const __bf16* p1) {
    v4u a = *(const v4u*)p0;
    v4u b = *(const v4u*)p1;
    v8u c;
    c[0]=a[0]; c[1]=a[1]; c[2]=a[2]; c[3]=a[3];
    c[4]=b[0]; c[5]=b[1]; c[6]=b[2]; c[7]=b[3];
    return __builtin_bit_cast(v16bf, c);
}

#define WMMA_BF16(A, B, C) \
    __builtin_amdgcn_wmma_f32_16x16x32_bf16(false, (A), false, (B), (short)0, (C), false, false)

// ---- gfx1250 async global->LDS copy (ASYNCcnt-tracked), with sync fallback ----
#if __has_builtin(__builtin_amdgcn_global_load_async_to_lds_b128)
#define HAS_ASYNC_LDS 1
typedef __attribute__((address_space(1))) v4i g_v4i;   // global int4
typedef __attribute__((address_space(3))) v4i l_v4i;   // LDS int4
static __device__ inline void async_copy16(void* lds, const void* g) {
    __builtin_amdgcn_global_load_async_to_lds_b128(
        (g_v4i*)g, (l_v4i*)lds, 0, 0);
}
static __device__ inline void wait_async() {
#if __has_builtin(__builtin_amdgcn_s_wait_asynccnt)
    __builtin_amdgcn_s_wait_asynccnt(0);
#else
    asm volatile("s_wait_asynccnt 0x0" ::: "memory");
#endif
}
#else
#define HAS_ASYNC_LDS 0
static __device__ inline void async_copy16(void* lds, const void* g) {
    *(v4u*)lds = *(const v4u*)g;     // sync fallback through VGPRs
}
static __device__ inline void wait_async() {}
#endif

// ---------------------------------------------------------------------------
// Kernel 0: f32 -> bf16 conversion (memory bound; enables pure-bf16
// async LDS staging in the GEMMs). Each thread converts 8 elements.
// ---------------------------------------------------------------------------
__global__ __launch_bounds__(256) void cvt_bf16_kernel(
    const float* __restrict__ s, __bf16* __restrict__ d, int n8)
{
    int i = blockIdx.x * 256 + threadIdx.x;
    if (i >= n8) return;
    const float4* p = (const float4*)s + (size_t)i * 2;
    float4 a = p[0], b = p[1];
    union { v4u u; __bf16 h[8]; } pk;
    pk.h[0] = (__bf16)a.x; pk.h[1] = (__bf16)a.y;
    pk.h[2] = (__bf16)a.z; pk.h[3] = (__bf16)a.w;
    pk.h[4] = (__bf16)b.x; pk.h[5] = (__bf16)b.y;
    pk.h[6] = (__bf16)b.z; pk.h[7] = (__bf16)b.w;
    *((v4u*)d + i) = pk.u;
}

// ---------------------------------------------------------------------------
// Kernel 1: QKV projections (bf16 in, bf16 out, f32 accum).
// grid = (DIM/64, SEQ/128, 3), block = 256 (8 waves).
// Double-buffered LDS, async global->LDS staging, one barrier per k-step.
// mat==1 (K) stored transposed: Kt[d][seq].
// ---------------------------------------------------------------------------
__global__ __launch_bounds__(256) void qkv_gemm_kernel(
    const __bf16* __restrict__ Xb,
    const __bf16* __restrict__ Wqb,
    const __bf16* __restrict__ Wkb,
    const __bf16* __restrict__ Wvb,
    __bf16* __restrict__ Qb,
    __bf16* __restrict__ Ktb,
    __bf16* __restrict__ Vb)
{
    __shared__ __bf16 As[2][128 * 32];   // 2 x 8 KB
    __shared__ __bf16 Bs[2][32 * 64];    // 2 x 4 KB

    const int tid  = threadIdx.x;
    const int wave = tid >> 5;
    const int lane = tid & 31;
    const int mblk = blockIdx.y * 128;
    const int nblk = blockIdx.x * 64;
    const int mat  = blockIdx.z;
    const __bf16* Wb = (mat == 0) ? Wqb : (mat == 1) ? Wkb : Wvb;

    // stage one 128x32 X tile + 32x64 W tile into LDS buffer `buf`
    auto stage = [&](int buf, int kk) {
        // As: 512 x 16B chunks, 2 per thread (row = 4 chunks of 8 bf16)
        for (int i = 0; i < 2; ++i) {
            int q   = tid + i * 256;
            int row = q >> 2;
            int c   = (q & 3) * 8;
            async_copy16(&As[buf][row * 32 + c],
                         Xb + (size_t)(mblk + row) * DIM + kk + c);
        }
        // Bs: 256 x 16B chunks, 1 per thread (row = 8 chunks of 8 bf16)
        {
            int row = tid >> 3;
            int c   = (tid & 7) * 8;
            async_copy16(&Bs[buf][row * 64 + c],
                         Wb + (size_t)(kk + row) * DIM + nblk + c);
        }
    };

    v8f acc[4];
    for (int i = 0; i < 4; ++i) acc[i] = vzero8();

    stage(0, 0);
    wait_async();
    __syncthreads();

    int buf = 0;
    const int arow  = wave * 16 + (lane & 15);
    const int ahalf = (lane >> 4) * 8;

    for (int kk = 0; kk < DIM; kk += 32) {
        if (kk + 32 < DIM) stage(buf ^ 1, kk + 32);   // prefetch next tile

        v16bf a = load_frag2(&As[buf][arow * 32 + ahalf],
                             &As[buf][arow * 32 + 16 + ahalf]);
        for (int nt = 0; nt < 4; ++nt) {
            v16bf b;
            for (int s2 = 0; s2 < 16; ++s2)
                b[s2] = Bs[buf][((s2 & 7) + ((s2 >> 3) << 4) + ahalf) * 64 + nt * 16 + (lane & 15)];
            acc[nt] = WMMA_BF16(a, b, acc[nt]);
        }

        wait_async();        // next buffer landed
        __syncthreads();     // everyone done reading current buffer
        buf ^= 1;
    }

    // store (C/D layout: vgpr r, lane<16 -> M=r, lane>=16 -> M=8+r)
    const int mrow = mblk + wave * 16 + 8 * (lane >> 4);
    if (mat == 1) {
        for (int nt = 0; nt < 4; ++nt) {
            int n = nblk + nt * 16 + (lane & 15);
            union { v4u u; __bf16 h[8]; } pk;
            for (int r = 0; r < 8; ++r) pk.h[r] = (__bf16)acc[nt][r];
            *(v4u*)(&Ktb[(size_t)n * SEQ + mrow]) = pk.u;
        }
    } else {
        __bf16* O = (mat == 0) ? Qb : Vb;
        for (int nt = 0; nt < 4; ++nt) {
            int n = nblk + nt * 16 + (lane & 15);
            for (int r = 0; r < 8; ++r)
                O[(size_t)(mrow + r) * DIM + n] = (__bf16)acc[nt][r];
        }
    }
}

// ---------------------------------------------------------------------------
// Kernel 2: flash attention. grid = SEQ/16 blocks, block = 256 (8 waves).
// Block owns 16 query rows; wave w owns output cols [128w,128w+128).
// Online softmax over key blocks of 128; P staged in LDS as bf16.
// Kt/V streamed from global (16MB bf16 total -> resident in 192MB L2).
// ---------------------------------------------------------------------------
__global__ __launch_bounds__(256) void flash_attn_kernel(
    const __bf16* __restrict__ Qb,
    const __bf16* __restrict__ Ktb,   // [DIM][SEQ]
    const __bf16* __restrict__ Vb,    // [SEQ][DIM]
    float* __restrict__ Out)
{
    __shared__ __bf16 Qs[16 * DIM];   // 32 KB, resident all kernel
    __shared__ __bf16 Ps[16 * 128];   // 4 KB probability tile
    __shared__ float  red[8 * 16];
    __shared__ float  mstate[16], lstate[16], sfac[16], mnew[16];

    const int tid   = threadIdx.x;
    const int wave  = tid >> 5;
    const int lane  = tid & 31;
    const int qbase = blockIdx.x * 16;
    const int c0    = wave * 128;

    // async-stage 16 Q rows (2048 x 16B chunks, 8 per thread)
    for (int i = 0; i < 8; ++i) {
        int q   = tid + i * 256;
        int row = q >> 7;
        int c   = (q & 127) * 8;
        async_copy16(&Qs[row * DIM + c],
                     Qb + (size_t)(qbase + row) * DIM + c);
    }
    if (tid < 16) { mstate[tid] = -1e30f; lstate[tid] = 0.f; }

    v8f acc[8];
    for (int i = 0; i < 8; ++i) acc[i] = vzero8();

    const int rowoff = (lane >> 4) * 8;
    wait_async();
    __syncthreads();

    for (int j = 0; j < 32; ++j) {
        // warm L2/L0 for next key block (global_prefetch_b8)
        if (j + 1 < 32) {
            __builtin_prefetch(Ktb + (size_t)lane * SEQ + (j + 1) * 128 + wave * 16, 0, 1);
            __builtin_prefetch(Vb + (size_t)((j + 1) * 128 + lane) * DIM + c0, 0, 1);
        }

        // ---- Phase A: scores. wave computes S[16x16] for keys kb..kb+15 ----
        const int kb = j * 128 + wave * 16;
        v8f s = vzero8();
        for (int d = 0; d < DIM; d += 32) {
            int arow = lane & 15;
            v16bf a = load_frag2(&Qs[arow * DIM + d + rowoff],
                                 &Qs[arow * DIM + d + 16 + rowoff]);
            v16bf b;
            for (int s2 = 0; s2 < 16; ++s2)
                b[s2] = Ktb[(size_t)(d + (s2 & 7) + ((s2 >> 3) << 4) + rowoff) * SEQ + kb + (lane & 15)];
            s = WMMA_BF16(a, b, s);
        }
        for (int r = 0; r < 8; ++r) s[r] *= 0.03125f;   // 1/sqrt(1024)

        // per-row max across the 16 lanes of each half-wave
        float rm[8];
        for (int r = 0; r < 8; ++r) rm[r] = s[r];
        for (int off = 1; off < 16; off <<= 1)
            for (int r = 0; r < 8; ++r)
                rm[r] = fmaxf(rm[r], __shfl_xor(rm[r], off, 32));
        if ((lane & 15) == 0)
            for (int r = 0; r < 8; ++r) red[wave * 16 + rowoff + r] = rm[r];
        __syncthreads();

        // combine maxima across waves, update running softmax state
        if (tid < 16) {
            float m  = mstate[tid];
            float mb = red[tid];
            for (int w = 1; w < 8; ++w) mb = fmaxf(mb, red[w * 16 + tid]);
            float mn = fmaxf(m, mb);
            float f  = __expf(m - mn);
            sfac[tid]   = f;
            mnew[tid]   = mn;
            mstate[tid] = mn;
            lstate[tid] *= f;
        }
        __syncthreads();

        // P = exp(S - mnew) -> LDS bf16; per-wave row sums
        float rs[8];
        const int keyl = wave * 16 + (lane & 15);
        for (int r = 0; r < 8; ++r) {
            int row = rowoff + r;
            float p = __expf(s[r] - mnew[row]);
            Ps[row * 128 + keyl] = (__bf16)p;
            rs[r] = p;
        }
        for (int off = 1; off < 16; off <<= 1)
            for (int r = 0; r < 8; ++r) rs[r] += __shfl_xor(rs[r], off, 32);
        if ((lane & 15) == 0)
            for (int r = 0; r < 8; ++r) red[wave * 16 + rowoff + r] = rs[r];

        // rescale output accumulators by exp(m_old - m_new)
        float f8[8];
        for (int r = 0; r < 8; ++r) f8[r] = sfac[rowoff + r];
        for (int nt = 0; nt < 8; ++nt)
            for (int r = 0; r < 8; ++r) acc[nt][r] *= f8[r];
        __syncthreads();

        if (tid < 16) {
            float t = 0.f;
            for (int w = 0; w < 8; ++w) t += red[w * 16 + tid];
            lstate[tid] += t;
        }

        // ---- Phase B: acc += P[16x128] @ V[128 x 128cols] for this wave ----
        for (int kkk = 0; kkk < 128; kkk += 32) {
            int arow = lane & 15;
            v16bf a = load_frag2(&Ps[arow * 128 + kkk + rowoff],
                                 &Ps[arow * 128 + kkk + 16 + rowoff]);
            const __bf16* vrow = Vb + (size_t)(j * 128 + kkk) * DIM + c0;
            for (int nt = 0; nt < 8; ++nt) {
                v16bf b;
                for (int s2 = 0; s2 < 16; ++s2)
                    b[s2] = vrow[(size_t)((s2 & 7) + ((s2 >> 3) << 4) + rowoff) * DIM + nt * 16 + (lane & 15)];
                acc[nt] = WMMA_BF16(a, b, acc[nt]);
            }
        }
        __syncthreads();
    }

    // normalize by row sums and write f32 output
    float inv[8];
    for (int r = 0; r < 8; ++r) inv[r] = 1.f / lstate[rowoff + r];
    for (int nt = 0; nt < 8; ++nt) {
        int col = c0 + nt * 16 + (lane & 15);
        for (int r = 0; r < 8; ++r) {
            int row = qbase + rowoff + r;
            Out[(size_t)row * DIM + col] = acc[nt][r] * inv[r];
        }
    }
}

// ---------------------------------------------------------------------------
extern "C" void kernel_launch(void* const* d_in, const int* in_sizes, int n_in,
                              void* d_out, int out_size, void* d_ws, size_t ws_size,
                              hipStream_t stream) {
    const float* X  = (const float*)d_in[0];
    const float* Wq = (const float*)d_in[1];
    const float* Wk = (const float*)d_in[2];
    const float* Wv = (const float*)d_in[3];
    float* Out = (float*)d_out;

    const size_t seqMat = (size_t)SEQ * DIM * sizeof(__bf16);   // 8 MB
    const size_t wMat   = (size_t)DIM * DIM * sizeof(__bf16);   // 2 MB
    if (ws_size < 4 * seqMat + 3 * wMat) return;                // 38 MB

    char* ws = (char*)d_ws;
    __bf16* Qb  = (__bf16*)(ws);
    __bf16* Ktb = (__bf16*)(ws + seqMat);
    __bf16* Vb  = (__bf16*)(ws + 2 * seqMat);
    __bf16* Xb  = (__bf16*)(ws + 3 * seqMat);
    __bf16* Wqb = (__bf16*)(ws + 4 * seqMat);
    __bf16* Wkb = (__bf16*)(ws + 4 * seqMat + wMat);
    __bf16* Wvb = (__bf16*)(ws + 4 * seqMat + 2 * wMat);

    // f32 -> bf16 conversion passes (memory bound)
    const int nX8 = SEQ * DIM / 8, nW8 = DIM * DIM / 8;
    cvt_bf16_kernel<<<dim3((nX8 + 255) / 256), 256, 0, stream>>>(X, Xb, nX8);
    cvt_bf16_kernel<<<dim3((nW8 + 255) / 256), 256, 0, stream>>>(Wq, Wqb, nW8);
    cvt_bf16_kernel<<<dim3((nW8 + 255) / 256), 256, 0, stream>>>(Wk, Wkb, nW8);
    cvt_bf16_kernel<<<dim3((nW8 + 255) / 256), 256, 0, stream>>>(Wv, Wvb, nW8);

    // QKV projections: 3 GEMMs of 4096x1024 @ 1024x1024 (bf16 WMMA, f32 accum)
    qkv_gemm_kernel<<<dim3(DIM / 64, SEQ / 128, 3), 256, 0, stream>>>(
        Xb, Wqb, Wkb, Wvb, Qb, Ktb, Vb);

    // fused flash attention: softmax(QK^T/32) @ V
    flash_attn_kernel<<<dim3(SEQ / 16), 256, 0, stream>>>(Qb, Ktb, Vb, Out);
}
